// DegModel_58119497449653
// MI455X (gfx1250) — matrix operand order repaired and checked by reading, not compiled, not run on hardware
//
#include <hip/hip_runtime.h>

typedef __attribute__((ext_vector_type(16))) __bf16 v16bf;
typedef __attribute__((ext_vector_type(8)))  float  v8f;

#define NB_   8
#define NF_   64
#define HZ_   64
#define WZ_   64
#define NPIX  (HZ_*WZ_)      // 4096
#define K2_   441
#define KS_   21
#define BATCH 8
#define HX    256
#define WX    256
#define RES_FRAG_PER_LAYER (4*18*32*16)   // 36864 bf16 per conv layer

static __device__ __forceinline__ v8f zero8() {
  v8f z = {0.f,0.f,0.f,0.f,0.f,0.f,0.f,0.f};
  return z;
}

// ---------------------------------------------------------------------------
// Pack 64->64 3x3 weights (OIHW f32) into bf16 WMMA A-fragments.
// K ordering: K = tap*64 + ic  (tap = ky*3+kx). Per k-step s (32 wide):
// tap = s/2, ic base = (s&1)*32. A-frag k order used here (16x32): lane l holds row
// M = l&15; element e holds K_local = 16*(l>>4) + e.
// Fragment storage: wfrag[((L*4 + mt)*18 + s)*32 + l][e]
// ---------------------------------------------------------------------------
__global__ void prep_wfrag_res(const float* __restrict__ rw1,
                               const float* __restrict__ rw2,
                               __bf16* __restrict__ wfrag) {
  int idx = blockIdx.x * blockDim.x + threadIdx.x;
  if (idx >= 16 * RES_FRAG_PER_LAYER) return;
  int e = idx & 15;
  int t = idx >> 4;
  int l = t & 31; t >>= 5;
  int s = t % 18; t /= 18;
  int mt = t & 3;
  int L  = t >> 2;                 // conv layer 0..15 (block i: 2i, 2i+1)
  int blk = L >> 1;
  const float* src = (L & 1) ? (rw2 + (size_t)blk * NF_ * NF_ * 9)
                             : (rw1 + (size_t)blk * NF_ * NF_ * 9);
  int oc  = mt * 16 + (l & 15);
  int tap = s >> 1, ky = tap / 3, kx = tap % 3;
  int ic  = ((s & 1) << 5) + ((l >> 4) << 4) + e;
  wfrag[idx] = (__bf16)src[((oc * NF_ + ic) * 3 + ky) * 3 + kx];
}

// last conv: 1x1, M=441 padded to 448 (28 tiles), K=64 (2 k-steps)
__global__ void prep_wfrag_last(const float* __restrict__ lw,
                                __bf16* __restrict__ wfrag) {
  int idx = blockIdx.x * blockDim.x + threadIdx.x;
  if (idx >= 28 * 2 * 32 * 16) return;
  int e = idx & 15;
  int t = idx >> 4;
  int l = t & 31; t >>= 5;
  int s = t & 1;
  int mt = t >> 1;
  int oc = mt * 16 + (l & 15);
  int ic = (s << 5) + ((l >> 4) << 4) + e;
  float v = (oc < K2_) ? lw[oc * NF_ + ic] : 0.f;
  wfrag[idx] = (__bf16)v;
}

// ---------------------------------------------------------------------------
// conv0: 3->64 3x3 SAME on z (B,3,64,64). Only 0.3% of total FLOPs -> VALU.
// ---------------------------------------------------------------------------
__global__ void conv0_k(const float* __restrict__ z, const float* __restrict__ w,
                        const float* __restrict__ bias, float* __restrict__ h) {
  int idx = blockIdx.x * blockDim.x + threadIdx.x;
  if (idx >= BATCH * NF_ * NPIX) return;
  int p = idx & (NPIX - 1);
  int t = idx >> 12;
  int oc = t & 63;
  int b  = t >> 6;
  int y = p >> 6, x = p & 63;
  float acc = bias[oc];
  #pragma unroll
  for (int ic = 0; ic < 3; ++ic)
    #pragma unroll
    for (int ky = 0; ky < 3; ++ky) {
      int yy = y + ky - 1;
      if (yy < 0 || yy >= HZ_) continue;
      #pragma unroll
      for (int kx = 0; kx < 3; ++kx) {
        int xx = x + kx - 1;
        if (xx < 0 || xx >= WZ_) continue;
        acc += z[((b * 3 + ic) * HZ_ + yy) * WZ_ + xx] *
               w[((oc * 3 + ic) * 3 + ky) * 3 + kx];
      }
    }
  h[idx] = acc;
}

// ---------------------------------------------------------------------------
// 64->64 3x3 SAME conv as implicit GEMM via bf16 WMMA, f32 accumulation.
// Block: (half-row, row, batch), 128 threads = 4 waves. Wave w: M-tile w
// (16 oc) x 2 N-tiles (32 px). LDS tile: 3 rows x 34 cols x 64 ic bf16.
// mode 0: out = relu(conv+bias)   mode 1: out = resid + (conv+bias)
// ---------------------------------------------------------------------------
__global__ __launch_bounds__(128) void resconv_wmma(
    const float* __restrict__ in, const __bf16* __restrict__ wfrag,
    const float* __restrict__ bias, const float* __restrict__ resid,
    float* __restrict__ out, int mode) {
  __shared__ __align__(32) __bf16 lds[3 * 34 * NF_];
  int tid = threadIdx.x;
  int lane = tid & 31, wave = tid >> 5;
  int b = blockIdx.z, y = blockIdx.y, xbase = blockIdx.x * 32;

  const float* inb = in + (size_t)b * NF_ * NPIX;
  for (int j = tid; j < 3 * 34 * NF_; j += 128) {
    int ic  = j / (3 * 34);
    int rem = j - ic * (3 * 34);
    int r = rem / 34, c = rem - r * 34;
    int yy = y + r - 1, xx = xbase + c - 1;
    float v = 0.f;
    if (yy >= 0 && yy < HZ_ && xx >= 0 && xx < WZ_)
      v = inb[ic * NPIX + yy * WZ_ + xx];
    lds[(r * 34 + c) * NF_ + ic] = (__bf16)v;
  }
  __syncthreads();

  int mt = wave;
  int hi = lane >> 4;
  v8f acc0 = zero8(), acc1 = zero8();
  for (int s = 0; s < 18; ++s) {
    int tap = s >> 1, ky = tap / 3, kx = tap - 3 * (tap / 3);
    int cb = ((s & 1) << 5) + (hi << 4);              // B frag: K = 16*(l>>4)+e
    v16bf a = *(const v16bf*)(wfrag + (size_t)((mt * 18 + s) * 32 + lane) * 16);
    {
      int col = (lane & 15) + kx;
      v16bf bf = *(const v16bf*)(&lds[(ky * 34 + col) * NF_ + cb]);
      acc0 = __builtin_amdgcn_wmma_f32_16x16x32_bf16(false, a, false, bf,
                                                     (short)0, acc0, false, false);
    }
    {
      int col = 16 + (lane & 15) + kx;
      v16bf bf = *(const v16bf*)(&lds[(ky * 34 + col) * NF_ + cb]);
      acc1 = __builtin_amdgcn_wmma_f32_16x16x32_bf16(false, a, false, bf,
                                                     (short)0, acc1, false, false);
    }
  }

  #pragma unroll
  for (int nt = 0; nt < 2; ++nt) {
    v8f acc = nt ? acc1 : acc0;
    int xg = xbase + nt * 16 + (lane & 15);
    #pragma unroll
    for (int r = 0; r < 8; ++r) {
      int oc = mt * 16 + r + (hi << 3);               // D: M = r + 8*(l>>4)
      float v = acc[r] + bias[oc];
      size_t o = (size_t)b * NF_ * NPIX + (size_t)oc * NPIX + y * WZ_ + xg;
      if (mode == 0) v = v > 0.f ? v : 0.f;
      else           v += resid[o];
      out[o] = v;
    }
  }
}

// ---------------------------------------------------------------------------
// last conv 1x1: 64 -> 441 (padded 448) via WMMA; writes raw kernel logits
// straight into d_out's kernel region, bias fused.
// ---------------------------------------------------------------------------
__global__ __launch_bounds__(128) void lastconv_wmma(
    const float* __restrict__ in, const __bf16* __restrict__ wfrag,
    const float* __restrict__ bias, float* __restrict__ kout) {
  __shared__ __align__(32) __bf16 lds[32 * NF_];
  int tid = threadIdx.x, lane = tid & 31, wave = tid >> 5;
  int b = blockIdx.z, y = blockIdx.y, xbase = blockIdx.x * 32;
  const float* inb = in + (size_t)b * NF_ * NPIX;
  for (int j = tid; j < 32 * NF_; j += 128) {
    int ic = j >> 5, px = j & 31;
    lds[px * NF_ + ic] = (__bf16)inb[ic * NPIX + y * WZ_ + xbase + px];
  }
  __syncthreads();

  int hi = lane >> 4;
  for (int mt = wave; mt < 28; mt += 4) {
    v8f acc0 = zero8(), acc1 = zero8();
    #pragma unroll
    for (int s = 0; s < 2; ++s) {
      int cb = (s << 5) + (hi << 4);
      v16bf a = *(const v16bf*)(wfrag + (size_t)((mt * 2 + s) * 32 + lane) * 16);
      v16bf b0 = *(const v16bf*)(&lds[(lane & 15) * NF_ + cb]);
      acc0 = __builtin_amdgcn_wmma_f32_16x16x32_bf16(false, a, false, b0,
                                                     (short)0, acc0, false, false);
      v16bf b1 = *(const v16bf*)(&lds[(16 + (lane & 15)) * NF_ + cb]);
      acc1 = __builtin_amdgcn_wmma_f32_16x16x32_bf16(false, a, false, b1,
                                                     (short)0, acc1, false, false);
    }
    #pragma unroll
    for (int nt = 0; nt < 2; ++nt) {
      v8f acc = nt ? acc1 : acc0;
      int xg = xbase + nt * 16 + (lane & 15);
      #pragma unroll
      for (int r = 0; r < 8; ++r) {
        int oc = mt * 16 + r + (hi << 3);
        if (oc < K2_)
          kout[((size_t)b * K2_ + oc) * NPIX + y * WZ_ + xg] = acc[r] + bias[oc];
      }
    }
  }
}

// normalize over the 441 taps, in place (coalesced across pixels per step)
__global__ void normalize_k(float* __restrict__ kout) {
  int idx = blockIdx.x * blockDim.x + threadIdx.x;
  if (idx >= BATCH * NPIX) return;
  int b = idx >> 12, p = idx & (NPIX - 1);
  float* base = kout + (size_t)b * K2_ * NPIX + p;
  float s = 0.f;
  for (int k = 0; k < K2_; ++k) s += base[(size_t)k * NPIX];
  float inv = 1.0f / (s + 1e-8f);
  for (int k = 0; k < K2_; ++k) base[(size_t)k * NPIX] *= inv;
}

// apply spatially-varying kernel: reflect-pad gather, stride 4
__global__ void apply_k(const float* __restrict__ x, const float* __restrict__ kn,
                        float* __restrict__ out) {
  int idx = blockIdx.x * blockDim.x + threadIdx.x;
  if (idx >= BATCH * 3 * NPIX) return;
  int p = idx & (NPIX - 1);
  int t = idx >> 12;
  int c = t % 3, b = t / 3;
  int i = p >> 6, j = p & 63;
  const float* xb = x + (size_t)(b * 3 + c) * HX * WX;
  const float* kb = kn + (size_t)b * K2_ * NPIX + p;
  float acc = 0.f;
  for (int ky = 0; ky < KS_; ++ky) {
    int r = 4 * i + ky - 10;
    r = r < 0 ? -r : (r > HX - 1 ? 2 * (HX - 1) - r : r);
    const float* xr = xb + (size_t)r * WX;
    for (int kx = 0; kx < KS_; ++kx) {
      int cc = 4 * j + kx - 10;
      cc = cc < 0 ? -cc : (cc > WX - 1 ? 2 * (WX - 1) - cc : cc);
      acc += kb[(size_t)(ky * KS_ + kx) * NPIX] * xr[cc];
    }
  }
  out[idx] = acc;
}

extern "C" void kernel_launch(void* const* d_in, const int* in_sizes, int n_in,
                              void* d_out, int out_size, void* d_ws, size_t ws_size,
                              hipStream_t stream) {
  (void)in_sizes; (void)n_in; (void)out_size; (void)ws_size;
  const float* x   = (const float*)d_in[0];
  const float* z   = (const float*)d_in[1];
  const float* c0w = (const float*)d_in[2];
  const float* c0b = (const float*)d_in[3];
  const float* rw1 = (const float*)d_in[4];
  const float* rb1 = (const float*)d_in[5];
  const float* rw2 = (const float*)d_in[6];
  const float* rb2 = (const float*)d_in[7];
  const float* lw  = (const float*)d_in[8];
  const float* lb  = (const float*)d_in[9];

  float* out  = (float*)d_out;                       // (B,3,64,64)
  float* kout = out + (size_t)BATCH * 3 * NPIX;      // (B,441,64,64)

  float*  h    = (float*)d_ws;                       // (B,64,64,64)  8 MB
  float*  tmp  = h + (size_t)BATCH * NF_ * NPIX;     // (B,64,64,64)  8 MB
  __bf16* wfr  = (__bf16*)(tmp + (size_t)BATCH * NF_ * NPIX); // 1.15 MB
  __bf16* wfl  = wfr + (size_t)16 * RES_FRAG_PER_LAYER;       // 56 KB

  prep_wfrag_res <<<16 * RES_FRAG_PER_LAYER / 256, 256, 0, stream>>>(rw1, rw2, wfr);
  prep_wfrag_last<<<28 * 2 * 32 * 16 / 256, 256, 0, stream>>>(lw, wfl);
  conv0_k<<<BATCH * NF_ * NPIX / 256, 256, 0, stream>>>(z, c0w, c0b, h);

  dim3 grid(2, HZ_, BATCH);   // (half-row, row, batch)
  for (int i = 0; i < NB_; ++i) {
    resconv_wmma<<<grid, 128, 0, stream>>>(
        h, wfr + (size_t)(2 * i) * RES_FRAG_PER_LAYER, rb1 + i * NF_,
        nullptr, tmp, 0);
    resconv_wmma<<<grid, 128, 0, stream>>>(
        tmp, wfr + (size_t)(2 * i + 1) * RES_FRAG_PER_LAYER, rb2 + i * NF_,
        h, h, 1);
  }
  lastconv_wmma<<<grid, 128, 0, stream>>>(h, wfl, lb, kout);
  normalize_k<<<BATCH * NPIX / 256, 256, 0, stream>>>(kout);
  apply_k<<<BATCH * 3 * NPIX / 256, 256, 0, stream>>>(x, kout, out);
}
